// DebertaRoPESelfAttention_33139967656375
// MI455X (gfx1250) — matrix operand order repaired and checked by reading, not compiled, not run on hardware
//
#include <hip/hip_runtime.h>
#include <hip/hip_bf16.h>

// ---------------------------------------------------------------------------
// DeBERTa-style RoPE self-attention for MI455X (gfx1250, wave32, WMMA).
// B=4, S=1024, H=16, D=64, HID=1024.
// - bf16 WMMA (fp32 accum) for all matmuls; every bf16-source fragment load is 2 x b128 (fp32 P: 4 x b128).
// - One-time transpose/convert kernels stage tiles with the Tensor Data Mover
//   (tensor_load_to_lds + s_wait_tensorcnt), HW-padded LDS layout; async-LDS
//   loads as fallback when the TDM builtin is unavailable.
// ---------------------------------------------------------------------------

typedef __attribute__((ext_vector_type(16))) __bf16 v16bf;
typedef __attribute__((ext_vector_type(8)))  __bf16 v8bf;
typedef __attribute__((ext_vector_type(4)))  __bf16 v4bf;
typedef __attribute__((ext_vector_type(8)))  float  v8f;
typedef __attribute__((ext_vector_type(4)))  unsigned v4u;
typedef __attribute__((ext_vector_type(8)))  int    v8i;
typedef __attribute__((ext_vector_type(4)))  int    v4i;

#define WMMA_BF16(a, b, c) \
  __builtin_amdgcn_wmma_f32_16x16x32_bf16(false, (a), false, (b), (short)0, (c), false, false)

#define BATCH 4
#define SEQ   1024
#define NHEAD 16
#define HDIM  64
#define HID   1024
#define QKVLD 3072
#define NEGF  (-3.4028235e38f)

// ---- A fragment (bf16): 16x32, per CDNA5 16-bit A layout --------------------
__device__ __forceinline__ v16bf load_A_frag_h(const __bf16* A, int lda, int row,
                                               int k0, int lane) {
  const int m = lane & 15, half = lane >> 4;
  const __bf16* p = A + (size_t)(row + m) * lda + k0 + 8 * half;
  const v8bf lo = *(const v8bf*)(p);
  const v8bf hi = *(const v8bf*)(p + 16);
  return __builtin_shufflevector(lo, hi, 0, 1, 2, 3, 4, 5, 6, 7,
                                 8, 9, 10, 11, 12, 13, 14, 15);
}

// ---- A fragment (fp32 source: LDS softmax probs) ----------------------------
__device__ __forceinline__ v16bf load_A_frag_f(const float* A, int lda, int row,
                                               int k0, int lane) {
  const int m = lane & 15, half = lane >> 4;
  const float* p = A + (size_t)(row + m) * lda + k0 + 8 * half;
  const float4 f0 = *(const float4*)(p + 0);
  const float4 f1 = *(const float4*)(p + 4);
  const float4 f2 = *(const float4*)(p + 16);
  const float4 f3 = *(const float4*)(p + 20);
  v16bf a;
  a[0]  = (__bf16)f0.x; a[1]  = (__bf16)f0.y; a[2]  = (__bf16)f0.z; a[3]  = (__bf16)f0.w;
  a[4]  = (__bf16)f1.x; a[5]  = (__bf16)f1.y; a[6]  = (__bf16)f1.z; a[7]  = (__bf16)f1.w;
  a[8]  = (__bf16)f2.x; a[9]  = (__bf16)f2.y; a[10] = (__bf16)f2.z; a[11] = (__bf16)f2.w;
  a[12] = (__bf16)f3.x; a[13] = (__bf16)f3.y; a[14] = (__bf16)f3.z; a[15] = (__bf16)f3.w;
  return a;
}

// ---- B fragment from TRANSPOSED bf16 source BT[n][k] (table k order) --------
__device__ __forceinline__ v16bf load_Bt_frag_h(const __bf16* Bt, int ldt, int k0,
                                                int col0, int lane) {
  const __bf16* p = Bt + (size_t)(col0 + (lane & 15)) * ldt + k0 + ((lane >> 4) << 3);
  const v8bf lo = *(const v8bf*)(p);
  const v8bf hi = *(const v8bf*)(p + 16);
  return __builtin_shufflevector(lo, hi, 0, 1, 2, 3, 4, 5, 6, 7,
                                 8, 9, 10, 11, 12, 13, 14, 15);
}

__device__ __forceinline__ void wait_async0() {
  asm volatile("s_wait_asynccnt 0x0" ::: "memory");
}

// ---------------------------------------------------------------------------
// Tensor Data Mover: 2D tile load into LDS per ISA ch.8 D# descriptor.
// data_size_log: 0=1B,1=2B,2=4B. HW padding reproduces padded LDS tiles.
// ---------------------------------------------------------------------------
#define HAVE_TDM __has_builtin(__builtin_amdgcn_tensor_load_to_lds)
#if HAVE_TDM
__device__ __forceinline__ void tdm_load_2d(unsigned lds_addr, const void* gaddr,
                                            unsigned data_size_log,
                                            unsigned tile_d0, unsigned tile_d1,
                                            unsigned tensor_d0, unsigned tensor_d1,
                                            unsigned d0_stride,
                                            unsigned pad_interval,
                                            unsigned pad_amount,
                                            bool pad_en) {
  const unsigned long long ga = (unsigned long long)gaddr;
  v4u g0;
  g0[0] = 1u;                                    // count=1, user mode
  g0[1] = lds_addr;                              // lds_addr[31:0]
  g0[2] = (unsigned)(ga & 0xFFFFFFFFu);          // global_addr[31:0]
  g0[3] = (unsigned)((ga >> 32) & 0x01FFFFFFu)   // global_addr[56:32]
          | (2u << 30);                          // type = 2 ("image")
  v8i g1;
  g1[0] = (int)((data_size_log << 16) | (pad_en ? (1u << 20) : 0u) |
                (pad_interval << 22) | (pad_amount << 25));
  g1[1] = (int)((tensor_d0 & 0xFFFFu) << 16);                       // dim0[15:0]
  g1[2] = (int)(((tensor_d0 >> 16) & 0xFFFFu) |
                ((tensor_d1 & 0xFFFFu) << 16));                     // dim0 hi|dim1 lo
  g1[3] = (int)(((tensor_d1 >> 16) & 0xFFFFu) |
                ((tile_d0 & 0xFFFFu) << 16));                       // dim1 hi|tile0
  g1[4] = (int)(tile_d1 & 0xFFFFu);                                 // tile1, tile2=0
  g1[5] = (int)d0_stride;                                           // stride[31:0]
  g1[6] = 0;                                                        // stride hi, dim1_stride lo
  g1[7] = 0;
  const v4i z = {0, 0, 0, 0};
#if defined(__clang_major__) && (__clang_major__ >= 23)
  const v8i z8 = {0, 0, 0, 0, 0, 0, 0, 0};
  __builtin_amdgcn_tensor_load_to_lds(g0, g1, z, z, z8, 0);
#else
  __builtin_amdgcn_tensor_load_to_lds(g0, g1, z, z, 0);
#endif
}
#endif

// ---------------------------------------------------------------------------
// GEMM: C[M,N] = A[M,K] @ BT^T[K,N] + bias[N]   (bf16 operands, fp32 accum)
// BT is [N][K] row-major. grid = (N/64, M/64), block = 128 (4 waves).
// ---------------------------------------------------------------------------
__global__ __launch_bounds__(128)
void gemm_wmma_h(const __bf16* __restrict__ A, const __bf16* __restrict__ BT,
                 const float* __restrict__ bias, float* __restrict__ C,
                 int lda, int ldbt, int ldc, int K) {
  const int lane = threadIdx.x & 31;
  const int wv   = threadIdx.x >> 5;
  const int row0 = blockIdx.y * 64 + wv * 16;
  const int col0 = blockIdx.x * 64;

  v8f acc[4] = {};
  for (int k0 = 0; k0 < K; k0 += 32) {
    if (k0 + 32 < K) {
      __builtin_prefetch(A + (size_t)(row0 + (lane & 15)) * lda + k0 + 32, 0, 1);
      __builtin_prefetch(BT + (size_t)(col0 + (lane & 15)) * ldbt + k0 + 32, 0, 1);
    }
    const v16bf a = load_A_frag_h(A, lda, row0, k0, lane);
#pragma unroll
    for (int t = 0; t < 4; ++t) {
      const v16bf b = load_Bt_frag_h(BT, ldbt, k0, col0 + 16 * t, lane);
      acc[t] = WMMA_BF16(a, b, acc[t]);
    }
  }

  const int n = lane & 15;
  const int mb = (lane >> 4) * 8;
#pragma unroll
  for (int t = 0; t < 4; ++t) {
    const float bv = bias[col0 + 16 * t + n];
    float* cp = C + (size_t)(row0 + mb) * ldc + col0 + 16 * t + n;
#pragma unroll
    for (int r = 0; r < 8; ++r) cp[(size_t)r * ldc] = acc[t][r] + bv;
  }
}

// ---------------------------------------------------------------------------
// fp32 -> bf16 plain conversion (n multiple of 1024); 4 elements/thread.
// ---------------------------------------------------------------------------
__global__ __launch_bounds__(256)
void cvt_bf16_kernel(const float* __restrict__ src, __bf16* __restrict__ dst) {
  const int i = (blockIdx.x * 256 + threadIdx.x) * 4;
  const float4 f = *(const float4*)(src + i);
  v4bf o;
  o[0] = (__bf16)f.x; o[1] = (__bf16)f.y; o[2] = (__bf16)f.z; o[3] = (__bf16)f.w;
  *(v4bf*)(dst + i) = o;
}

// ---------------------------------------------------------------------------
// One-time weight transpose+convert: src fp32 [R][C] -> dst bf16 [C][R].
// 64x64 tile staged by the TDM (HW pad: 4 DWORDs every 64 DWORDs -> [64][68]).
// grid = (C/64, R/64), block = 256.
// ---------------------------------------------------------------------------
__global__ __launch_bounds__(256)
void transpose_cvt_kernel(const float* __restrict__ src, __bf16* __restrict__ dst,
                          int src_ld, int src_rows, int src_cols, int dst_ld) {
  __shared__ float tile[64][68];
  const int tid = threadIdx.x;
  const int r0 = blockIdx.y * 64;
  const int c0 = blockIdx.x * 64;
  const unsigned tbase = (unsigned)(uintptr_t)&tile[0][0];

#if HAVE_TDM
  if (tid < 32) {                                // wave 0 issues one TDM op
    tdm_load_2d(tbase, src + (size_t)r0 * src_ld + c0,
                /*data_size_log=*/2, /*tile=*/64, 64,
                /*tensor=*/(unsigned)src_cols, (unsigned)src_rows,
                /*stride=*/(unsigned)src_ld,
                /*pad_interval: 64 DW=*/5, /*pad_amount: 4 DW=*/3, true);
    __builtin_amdgcn_s_wait_tensorcnt(0);
  }
#else
#pragma unroll
  for (int j = 0; j < 4; ++j) {
    const int ch = tid + j * 256;
    const int r = ch >> 4;
    const int cc = (ch & 15) * 4;
    const unsigned lds_off = tbase + (unsigned)((r * 68 + cc) * 4);
    const unsigned gofs =
        (unsigned)(((size_t)(r0 + r) * (size_t)src_ld + (size_t)(c0 + cc)) * 4u);
    asm volatile("global_load_async_to_lds_b128 %0, %1, %2"
                 :: "v"(lds_off), "v"(gofs), "s"(src)
                 : "memory");
  }
  wait_async0();
#endif
  __syncthreads();

  const int orow = tid & 63;                     // src col
  const int ch = tid >> 6;                       // 0..3
  __bf16* dp = dst + (size_t)(c0 + orow) * dst_ld + r0 + ch * 16;
  v8bf o0, o1;
#pragma unroll
  for (int i = 0; i < 8; ++i) o0[i] = (__bf16)tile[ch * 16 + i][orow];
#pragma unroll
  for (int i = 0; i < 8; ++i) o1[i] = (__bf16)tile[ch * 16 + 8 + i][orow];
  *(v8bf*)(dp) = o0;
  *(v8bf*)(dp + 8) = o1;
}

// ---------------------------------------------------------------------------
// Per-(b,h) V transpose: v rows [S][D] (bf16, stride QKVLD inside qkvh)
//   -> vT [B,H,D,S]. TDM-staged (HW pad: 4 DWORDs every 32 DWORDs -> [64][72]).
// grid = (S/64, B*H), block = 256.
// ---------------------------------------------------------------------------
__global__ __launch_bounds__(256)
void transpose_v_kernel(const __bf16* __restrict__ qkvh, __bf16* __restrict__ vT) {
  __shared__ __bf16 tile[64][72];
  const int tid = threadIdx.x;
  const int bh = blockIdx.y;
  const int b = bh >> 4, h = bh & 15;
  const int s0 = blockIdx.x * 64;
  const __bf16* src = qkvh + (size_t)b * SEQ * QKVLD + 2 * HID + h * HDIM;
  const unsigned tbase = (unsigned)(uintptr_t)&tile[0][0];

#if HAVE_TDM
  if (tid < 32) {
    tdm_load_2d(tbase, src + (size_t)s0 * QKVLD,
                /*data_size_log=*/1, /*tile=*/64, 64,
                /*tensor=*/64u, (unsigned)SEQ,
                /*stride=*/(unsigned)QKVLD,
                /*pad_interval: 32 DW=*/4, /*pad_amount: 4 DW=*/3, true);
    __builtin_amdgcn_s_wait_tensorcnt(0);
  }
#else
#pragma unroll
  for (int j = 0; j < 2; ++j) {
    const int ch = tid + j * 256;
    const int r = ch >> 3;
    const int cc = (ch & 7) * 8;
    const unsigned lds_off = tbase + (unsigned)((r * 72 + cc) * 2);
    const unsigned gofs =
        (unsigned)(((size_t)(s0 + r) * (size_t)QKVLD + (size_t)cc) * 2u);
    asm volatile("global_load_async_to_lds_b128 %0, %1, %2"
                 :: "v"(lds_off), "v"(gofs), "s"(src)
                 : "memory");
  }
  wait_async0();
#endif
  __syncthreads();

  const int od = tid & 63;                       // dim
  const int ch = tid >> 6;                       // 0..3 (s chunks of 16)
  __bf16* dp = vT + ((size_t)bh * HDIM + od) * SEQ + s0 + ch * 16;
  v8bf o0, o1;
#pragma unroll
  for (int i = 0; i < 8; ++i) o0[i] = tile[ch * 16 + i][od];
#pragma unroll
  for (int i = 0; i < 8; ++i) o1[i] = tile[ch * 16 + 8 + i][od];
  *(v8bf*)(dp) = o0;
  *(v8bf*)(dp + 8) = o1;
}

// ---------------------------------------------------------------------------
// RoPE: read fp32 qkv [B*S, 3*HID], rotate q/k pairs (j, j+32), emit bf16 copy.
// ---------------------------------------------------------------------------
__global__ __launch_bounds__(256)
void rope_kernel(const float* __restrict__ qkv, __bf16* __restrict__ qkvh) {
  const int idx = blockIdx.x * 256 + threadIdx.x;
  const int j = idx & 31;
  const int h = (idx >> 5) & 15;
  const int s = (idx >> 9) & (SEQ - 1);
  const int b = idx >> 19;

  const float f  = (float)s * __powf(10000.0f, -((float)(2 * j)) / 64.0f);
  const float cs = __cosf(f), sn = __sinf(f);

  const size_t base = ((size_t)(b * SEQ + s)) * QKVLD + h * HDIM + j;
  {  // q
    const float a = qkv[base], c = qkv[base + 32];
    qkvh[base]      = (__bf16)(a * cs - c * sn);
    qkvh[base + 32] = (__bf16)(c * cs + a * sn);
  }
  {  // k
    const float a = qkv[base + HID], c = qkv[base + HID + 32];
    qkvh[base + HID]      = (__bf16)(a * cs - c * sn);
    qkvh[base + HID + 32] = (__bf16)(c * cs + a * sn);
  }
  {  // v passthrough
    qkvh[base + 2 * HID]      = (__bf16)qkv[base + 2 * HID];
    qkvh[base + 2 * HID + 32] = (__bf16)qkv[base + 2 * HID + 32];
  }
}

// ---------------------------------------------------------------------------
// Attention: grid = (B*H, S/16); block = 128 (4 waves).
// ---------------------------------------------------------------------------
__global__ __launch_bounds__(128)
void attn_kernel(const __bf16* __restrict__ qkvh, const __bf16* __restrict__ vT,
                 const int* __restrict__ mask, __bf16* __restrict__ ctxh) {
  __shared__ float sc[16 * SEQ];                 // 64 KB

  const int bh = blockIdx.x;
  const int b  = bh >> 4;
  const int h  = bh & 15;
  const int row0 = blockIdx.y * 16;
  const int lane = threadIdx.x & 31;
  const int wv   = threadIdx.x >> 5;

  const __bf16* qbase = qkvh + (size_t)b * SEQ * QKVLD + (size_t)h * HDIM;
  const __bf16* kbase = qbase + HID;
  const __bf16* vTbh  = vT + (size_t)bh * HDIM * SEQ;   // [D][S]
  const int n  = lane & 15;
  const int mb = (lane >> 4) * 8;

  // ---- Phase 1: scores -> LDS (waves split the 64 key tiles) ----
  for (int kt = wv; kt < SEQ / 16; kt += 4) {
    v8f acc = {};
#pragma unroll
    for (int k0 = 0; k0 < HDIM; k0 += 32) {
      const v16bf a  = load_A_frag_h(qbase, QKVLD, row0, k0, lane);
      const v16bf kb = load_Bt_frag_h(kbase, QKVLD, k0, kt * 16, lane);
      acc = WMMA_BF16(a, kb, acc);
    }
    const int key = kt * 16 + n;
    const bool masked = (mask[b * SEQ + key] == 0);
#pragma unroll
    for (int r = 0; r < 8; ++r) {
      const float v = acc[r] * 0.125f;           // 1/sqrt(64)
      sc[(mb + r) * SEQ + key] = masked ? NEGF : v;
    }
  }
  __syncthreads();

  // ---- Phase 2: softmax (row = tid>>3; 8 lanes x 128-wide float4 chunks) ----
  {
    const int r = threadIdx.x >> 3;
    const int c = threadIdx.x & 7;
    float4* rp = (float4*)(sc + r * SEQ + c * 128);
    float mx = NEGF;
#pragma unroll 4
    for (int i = 0; i < 32; ++i) {
      const float4 v = rp[i];
      mx = fmaxf(mx, fmaxf(fmaxf(v.x, v.y), fmaxf(v.z, v.w)));
    }
    mx = fmaxf(mx, __shfl_xor(mx, 1, 32));
    mx = fmaxf(mx, __shfl_xor(mx, 2, 32));
    mx = fmaxf(mx, __shfl_xor(mx, 4, 32));
    float sum = 0.0f;
#pragma unroll 4
    for (int i = 0; i < 32; ++i) {
      float4 v = rp[i];
      v.x = __expf(v.x - mx); v.y = __expf(v.y - mx);
      v.z = __expf(v.z - mx); v.w = __expf(v.w - mx);
      sum += (v.x + v.y) + (v.z + v.w);
      rp[i] = v;
    }
    sum += __shfl_xor(sum, 1, 32);
    sum += __shfl_xor(sum, 2, 32);
    sum += __shfl_xor(sum, 4, 32);
    const float inv = 1.0f / sum;
#pragma unroll 4
    for (int i = 0; i < 32; ++i) {
      float4 v = rp[i];
      v.x *= inv; v.y *= inv; v.z *= inv; v.w *= inv;
      rp[i] = v;
    }
  }
  __syncthreads();

  // ---- Phase 3: ctx = P @ V  (wave wv owns dims [16*wv, 16*wv+16)) ----
  {
    v8f acc = {};
    for (int k0 = 0; k0 < SEQ; k0 += 32) {
      const v16bf a  = load_A_frag_f(sc, SEQ, 0, k0, lane);        // P from LDS
      const v16bf vb = load_Bt_frag_h(vTbh, SEQ, k0, wv * 16, lane);
      acc = WMMA_BF16(a, vb, acc);
    }
    __bf16* op = ctxh + (size_t)(b * SEQ + row0 + mb) * HID + h * HDIM + wv * 16 + n;
#pragma unroll
    for (int r = 0; r < 8; ++r) op[(size_t)r * HID] = (__bf16)acc[r];
  }
}

// ---------------------------------------------------------------------------
extern "C" void kernel_launch(void* const* d_in, const int* in_sizes, int n_in,
                              void* d_out, int out_size, void* d_ws, size_t ws_size,
                              hipStream_t stream) {
  const float* x     = (const float*)d_in[0];   // [B,S,HID]
  const int*   amask = (const int*)  d_in[1];   // [B,S]
  const float* Wqkv  = (const float*)d_in[2];   // [HID, 3*HID]
  const float* bqkv  = (const float*)d_in[3];   // [3*HID]
  const float* Wout  = (const float*)d_in[4];   // [HID, HID]
  const float* bout  = (const float*)d_in[5];   // [HID]
  float* out = (float*)d_out;                   // [B,S,HID]

  char* w = (char*)d_ws;
  float*  qkv    = (float*)w;  w += (size_t)BATCH * SEQ * QKVLD * 4;  // 48 MB
  __bf16* qkvh   = (__bf16*)w; w += (size_t)BATCH * SEQ * QKVLD * 2;  // 24 MB
  __bf16* xh     = (__bf16*)w; w += (size_t)BATCH * SEQ * HID * 2;    //  8 MB
  __bf16* wqkvT  = (__bf16*)w; w += (size_t)HID * QKVLD * 2;          //  6 MB
  __bf16* woutT  = (__bf16*)w; w += (size_t)HID * HID * 2;            //  2 MB
  __bf16* vT     = (__bf16*)w; w += (size_t)BATCH * NHEAD * HDIM * SEQ * 2; // 8 MB
  __bf16* ctxh   = (__bf16*)w; w += (size_t)BATCH * SEQ * HID * 2;    //  8 MB

  const dim3 blk(128);

  // 0) one-time operand prep: xh (cvt), WqkvT/WoutT (TDM transpose+cvt)
  cvt_bf16_kernel<<<(BATCH * SEQ * HID) / 1024, 256, 0, stream>>>(x, xh);
  transpose_cvt_kernel<<<dim3(QKVLD / 64, HID / 64), 256, 0, stream>>>(
      Wqkv, wqkvT, QKVLD, HID, QKVLD, HID);
  transpose_cvt_kernel<<<dim3(HID / 64, HID / 64), 256, 0, stream>>>(
      Wout, woutT, HID, HID, HID, HID);

  // 1) qkv = x @ Wqkv + bqkv        (M=4096, N=3072, K=1024)
  gemm_wmma_h<<<dim3(QKVLD / 64, (BATCH * SEQ) / 64), blk, 0, stream>>>(
      xh, wqkvT, bqkv, qkv, HID, HID, QKVLD, HID);

  // 2) RoPE q,k (fp32 in, bf16 out)
  rope_kernel<<<(BATCH * SEQ * NHEAD * 32) / 256, 256, 0, stream>>>(qkv, qkvh);

  // 3) vT [B,H,D,S] from qkvh (TDM-staged transpose)
  transpose_v_kernel<<<dim3(SEQ / 64, BATCH * NHEAD), 256, 0, stream>>>(qkvh, vT);

  // 4) attention -> ctxh [B*S, HID] bf16
  attn_kernel<<<dim3(BATCH * NHEAD, SEQ / 16), blk, 0, stream>>>(qkvh, vT, amask, ctxh);

  // 5) out = ctx @ Wout + bout      (M=4096, N=1024, K=1024)
  gemm_wmma_h<<<dim3(HID / 64, (BATCH * SEQ) / 64), blk, 0, stream>>>(
      ctxh, woutT, bout, out, HID, HID, HID, HID);
}
